// SwinTransformerBlock_44238163149005
// MI455X (gfx1250) — matrix operand order repaired and checked by reading, not compiled, not run on hardware
//
#include <hip/hip_runtime.h>
#include <hip/hip_bf16.h>

typedef _Float16 f16;
typedef __attribute__((ext_vector_type(16))) _Float16 v16h;
typedef __attribute__((ext_vector_type(8)))  float    v8f;
typedef int v4i __attribute__((vector_size(16)));

#define AS1 __attribute__((address_space(1)))
#define AS3 __attribute__((address_space(3)))

#define DIM     384
#define HEADS   12
#define QKVN    1152
#define MLP_HIDN 1536
#define NW      2048      // 32 batches * 8*8 windows
#define MTOT    100352    // 2048 * 49 tokens (window order) == 32*3136 (token order)

#if defined(__gfx1250__) && __has_builtin(__builtin_amdgcn_global_load_async_to_lds_b128)
#define HAVE_ASYNC_LDS 1
#else
#define HAVE_ASYNC_LDS 0
#endif

__device__ __forceinline__ void async_wait0() {
#if __has_builtin(__builtin_amdgcn_s_wait_asynccnt)
    __builtin_amdgcn_s_wait_asynccnt(0);
#else
    asm volatile("s_wait_asynccnt 0x0" ::: "memory");
#endif
}

// 16B global -> LDS copy; async (GLOBAL_LOAD_ASYNC_TO_LDS_B128, ASYNCcnt) when available
__device__ __forceinline__ void cp_async16(void* lds_dst, const void* gsrc) {
#if HAVE_ASYNC_LDS
    __builtin_amdgcn_global_load_async_to_lds_b128(
        (AS1 v4i*)(void*)gsrc,
        (AS3 v4i*)lds_dst, 0, 0);
#else
    *(uint4*)lds_dst = *(const uint4*)gsrc;
#endif
}

union FragU { v16h v; uint4 q[2]; };

// ---- WMMA fragment loaders (ISA 7.12.2 layouts), sources in LDS ----
// A: 16x32 f16, lane(0-15)=row M, half-wave selects K-substripes of 8.
__device__ __forceinline__ v16h frag_a(const f16* base, int stride_h, int lane) {
    int m = lane & 15, hi = lane >> 4;
    const char* r = (const char*)(base + m * stride_h) + hi * 16;
    FragU f;
    f.q[0] = *(const uint4*)(r);        // K = g*16 + hi*8 + {0..7}
    f.q[1] = *(const uint4*)(r + 32);
    return f.v;
}
// B: 32x16 f16, lane(0-15)=col N, half-wave selects K-substripes of 8.
__device__ __forceinline__ v16h frag_b(const f16* base, int stride_h, int lane) {
    int n = lane & 15, hi = lane >> 4;
    const char* r = (const char*)(base + n * stride_h) + hi * 16;
    FragU f;
    f.q[0] = *(const uint4*)(r);        // K = g*16 + hi*8 + {0..7}
    f.q[1] = *(const uint4*)(r + 32);
    return f.v;
}

__device__ __forceinline__ v8f wmma_f16(v16h a, v16h b, v8f c) {
    return __builtin_amdgcn_wmma_f32_16x16x32_f16(false, a, false, b, (short)0, c, false, false);
}

// window-order row r -> (batch, token index l); roll gather src == roll scatter dst (+3 mod 56)
__device__ __forceinline__ void win_map(int r, int& bb, int& l) {
    int n = r / 49, s = r - n * 49;
    bb = n >> 6;
    int rem = n & 63;
    int wh = rem >> 3, ww = rem & 7;
    int i = s / 7, j = s - i * 7;
    int y = wh * 7 + i + 3; if (y >= 56) y -= 56;
    int x = ww * 7 + j + 3; if (x >= 56) x -= 56;
    l = y * 56 + x;
}

// ---------------- LayerNorm kernels (one wave32 per token) ----------------
__global__ void ln1win_kernel(const float* __restrict__ x, const float* __restrict__ w,
                              const float* __restrict__ b, f16* __restrict__ act1) {
    int r    = blockIdx.x * 8 + (threadIdx.x >> 5);
    int lane = threadIdx.x & 31;
    int bb, l; win_map(r, bb, l);
    const float* row = x + ((size_t)bb * 3136 + l) * DIM;
    float vals[12], s1 = 0.f, s2 = 0.f;
#pragma unroll
    for (int t = 0; t < 12; ++t) { float v = row[t * 32 + lane]; vals[t] = v; s1 += v; s2 += v * v; }
#pragma unroll
    for (int off = 16; off > 0; off >>= 1) { s1 += __shfl_xor(s1, off); s2 += __shfl_xor(s2, off); }
    float mean = s1 * (1.f / DIM);
    float var  = s2 * (1.f / DIM) - mean * mean;
    float rstd = rsqrtf(var + 1e-5f);
    f16* orow = act1 + (size_t)r * DIM;
#pragma unroll
    for (int t = 0; t < 12; ++t) {
        int c = t * 32 + lane;
        orow[c] = (f16)(((vals[t] - mean) * rstd) * w[c] + b[c]);
    }
}

__global__ void ln2_kernel(const float* __restrict__ x2, const float* __restrict__ w,
                           const float* __restrict__ b, f16* __restrict__ h2) {
    int r    = blockIdx.x * 8 + (threadIdx.x >> 5);
    int lane = threadIdx.x & 31;
    const float* row = x2 + (size_t)r * DIM;
    float vals[12], s1 = 0.f, s2 = 0.f;
#pragma unroll
    for (int t = 0; t < 12; ++t) { float v = row[t * 32 + lane]; vals[t] = v; s1 += v; s2 += v * v; }
#pragma unroll
    for (int off = 16; off > 0; off >>= 1) { s1 += __shfl_xor(s1, off); s2 += __shfl_xor(s2, off); }
    float mean = s1 * (1.f / DIM);
    float var  = s2 * (1.f / DIM) - mean * mean;
    float rstd = rsqrtf(var + 1e-5f);
    f16* orow = h2 + (size_t)r * DIM;
#pragma unroll
    for (int t = 0; t < 12; ++t) {
        int c = t * 32 + lane;
        orow[c] = (f16)(((vals[t] - mean) * rstd) * w[c] + b[c]);
    }
}

__global__ void castw_kernel(const float* __restrict__ s, f16* __restrict__ d, int n) {
    int i = blockIdx.x * 256 + threadIdx.x;
    if (i < n) d[i] = (f16)s[i];
}

// ---------------- Tiled WMMA GEMM: out = A(MxK) * W(NxK)^T ----------------
// BM=128, BN=64, BK=64, 256 threads = 8 waves; wave owns one 16-row M tile x 4 N tiles.
// EPI 0: +bias, scale Q cols by 1/sqrt(32), store f16 (QKV)
// EPI 1: +bias + x[shortcut] via window scatter, store f32 x2 (out-proj)
// EPI 2: +bias, exact GELU, store f16 (fc1)
// EPI 3: +bias + x2, store f32 d_out (fc2)
template <int EPI>
__global__ void gemm_kernel(const f16* __restrict__ A, const f16* __restrict__ W,
                            const float* __restrict__ bias, int K, int N,
                            f16* __restrict__ hout, float* __restrict__ fout,
                            const float* __restrict__ addsrc) {
    __shared__ __align__(16) f16 sA[128 * 64];   // 16 KB
    __shared__ __align__(16) f16 sB[64 * 64];    //  8 KB
    int tid = threadIdx.x, wave = tid >> 5, lane = tid & 31;
    int m0 = blockIdx.x * 128, n0 = blockIdx.y * 64;
    v8f acc[4] = {};
    const uint4* Ag = (const uint4*)A;
    const uint4* Wg = (const uint4*)W;
    int kq = K >> 3;   // uint4 per row
    for (int kt = 0; kt < K; kt += 64) {
        int kc = kt >> 3;
        // stage A(128x64) + W(64x64) tiles: 1536 x 16B, 6 per thread
        for (int i = tid; i < 1536; i += 256) {
            if (i < 1024) {
                int r = i >> 3, c = i & 7;
                cp_async16(&((uint4*)sA)[i], &Ag[(size_t)(m0 + r) * kq + kc + c]);
            } else {
                int j = i - 1024;
                int r = j >> 3, c = j & 7;
                cp_async16(&((uint4*)sB)[j], &Wg[(size_t)(n0 + r) * kq + kc + c]);
            }
        }
        if (kt + 64 < K) {  // global_prefetch_b8 of next K-slab
            int pr = tid & 127;
            const void* p = (tid < 128) ? (const void*)&Ag[(size_t)(m0 + pr) * kq + kc + 8]
                                        : (const void*)&Wg[(size_t)(n0 + (pr & 63)) * kq + kc + 8];
            __builtin_prefetch(p, 0, 0);
        }
#if HAVE_ASYNC_LDS
        async_wait0();
#endif
        __syncthreads();
#pragma unroll
        for (int kk = 0; kk < 2; ++kk) {
            v16h af = frag_a(sA + wave * (16 * 64) + kk * 32, 64, lane);
#pragma unroll
            for (int nt = 0; nt < 4; ++nt) {
                v16h bf = frag_b(sB + nt * (16 * 64) + kk * 32, 64, lane);
                acc[nt] = wmma_f16(af, bf, acc[nt]);
            }
        }
        __syncthreads();
    }
    int hi = lane >> 4, ln = lane & 15;
#pragma unroll
    for (int v = 0; v < 8; ++v) {
        int row = m0 + wave * 16 + v + 8 * hi;
        int bb = 0, l = 0;
        if (EPI == 1) win_map(row, bb, l);
#pragma unroll
        for (int nt = 0; nt < 4; ++nt) {
            int col = n0 + nt * 16 + ln;
            float val = acc[nt][v] + bias[col];
            if (EPI == 0) {
                if (col < 384) val *= 0.17677669529663687f;   // 1/sqrt(hd)
                hout[(size_t)row * N + col] = (f16)val;
            } else if (EPI == 1) {
                size_t t = ((size_t)bb * 3136 + l) * 384 + col;
                fout[t] = val + addsrc[t];
            } else if (EPI == 2) {
                float g = 0.5f * val * (1.0f + erff(val * 0.70710678118654752f));
                hout[(size_t)row * N + col] = (f16)g;
            } else {
                size_t t = (size_t)row * 384 + col;
                fout[t] = val + addsrc[t];
            }
        }
    }
}

// ---------------- Attention: one block(4 waves) per (window, head); S=49 padded to 64 ----------------
__global__ void attn_kernel(const f16* __restrict__ qkv, f16* __restrict__ oc) {
    __shared__ __align__(16) f16   Qs[64 * 32];
    __shared__ __align__(16) f16   Ks[64 * 32];
    __shared__ __align__(16) f16   Vt[32 * 64];   // transposed: [dim][token]
    __shared__ __align__(16) float Lg[64 * 64];
    __shared__ __align__(16) f16   Ps[64 * 64];
    int widx = blockIdx.x, head = blockIdx.y;
    int tid = threadIdx.x, wave = tid >> 5, lane = tid & 31;
    const f16* base = qkv + (size_t)widx * 49 * QKVN;
    int q4 = (head * 32) >> 3, k4 = (384 + head * 32) >> 3, v4 = (768 + head * 32) >> 3;
    for (int i = tid; i < 256; i += 128) {
        int s = i >> 2, c = i & 3;
        if (s < 49) {
            const uint4* row = (const uint4*)(base + (size_t)s * QKVN);
            cp_async16(&((uint4*)Qs)[i], &row[q4 + c]);
            cp_async16(&((uint4*)Ks)[i], &row[k4 + c]);
            union { uint4 u; f16 h[8]; } t; t.u = row[v4 + c];
#pragma unroll
            for (int e = 0; e < 8; ++e) Vt[(c * 8 + e) * 64 + s] = t.h[e];
        } else {
            uint4 z = {0, 0, 0, 0};
            ((uint4*)Qs)[i] = z;
            ((uint4*)Ks)[i] = z;
#pragma unroll
            for (int e = 0; e < 8; ++e) Vt[(c * 8 + e) * 64 + s] = (f16)0.f;
        }
    }
#if HAVE_ASYNC_LDS
    async_wait0();
#endif
    __syncthreads();
    // logits = (scaled q) k^T : each wave owns one 16-row M tile
    {
        v16h aq = frag_a(Qs + wave * (16 * 32), 32, lane);
        int hi = lane >> 4, ln = lane & 15;
#pragma unroll
        for (int nt = 0; nt < 4; ++nt) {
            v8f c = {};
            v16h bk = frag_b(Ks + nt * (16 * 32), 32, lane);
            c = wmma_f16(aq, bk, c);
#pragma unroll
            for (int v = 0; v < 8; ++v)
                Lg[(wave * 16 + v + 8 * hi) * 64 + nt * 16 + ln] = c[v];
        }
    }
    __syncthreads();
    // fp32 softmax over 49 valid keys, one thread per row
    if (tid < 64) {
        float mx = -1e30f;
        for (int c = 0; c < 49; ++c) mx = fmaxf(mx, Lg[tid * 64 + c]);
        float s = 0.f;
        for (int c = 0; c < 49; ++c) { float e = __expf(Lg[tid * 64 + c] - mx); Lg[tid * 64 + c] = e; s += e; }
        float r = 1.f / s;
        for (int c = 0; c < 49; ++c) Ps[tid * 64 + c] = (f16)(Lg[tid * 64 + c] * r);
        for (int c = 49; c < 64; ++c) Ps[tid * 64 + c] = (f16)0.f;
    }
    __syncthreads();
    // O = P(64x64) * V(64x32); K dim in two 32-steps
    v8f oacc[2] = {};
#pragma unroll
    for (int kk = 0; kk < 2; ++kk) {
        v16h ap = frag_a(Ps + wave * (16 * 64) + kk * 32, 64, lane);
#pragma unroll
        for (int nt = 0; nt < 2; ++nt) {
            v16h bv = frag_b(Vt + nt * (16 * 64) + kk * 32, 64, lane);
            oacc[nt] = wmma_f16(ap, bv, oacc[nt]);
        }
    }
    int hi = lane >> 4, ln = lane & 15;
#pragma unroll
    for (int nt = 0; nt < 2; ++nt)
#pragma unroll
        for (int v = 0; v < 8; ++v) {
            int s = wave * 16 + v + 8 * hi;
            if (s < 49) {
                int d = nt * 16 + ln;
                oc[((size_t)widx * 49 + s) * DIM + head * 32 + d] = (f16)oacc[nt][v];
            }
        }
}

extern "C" void kernel_launch(void* const* d_in, const int* in_sizes, int n_in,
                              void* d_out, int out_size, void* d_ws, size_t ws_size,
                              hipStream_t stream) {
    (void)in_sizes; (void)n_in; (void)out_size; (void)ws_size;
    const float* x     = (const float*)d_in[0];
    const float* n1w   = (const float*)d_in[3];
    const float* n1b   = (const float*)d_in[4];
    const float* qkv_w = (const float*)d_in[5];
    const float* qkv_b = (const float*)d_in[6];
    const float* out_w = (const float*)d_in[7];
    const float* out_b = (const float*)d_in[8];
    const float* n2w   = (const float*)d_in[9];
    const float* n2b   = (const float*)d_in[10];
    const float* fc1_w = (const float*)d_in[11];
    const float* fc1_b = (const float*)d_in[12];
    const float* fc2_w = (const float*)d_in[13];
    const float* fc2_b = (const float*)d_in[14];
    float* out = (float*)d_out;

    // workspace layout (~543 MB, all segments 16B aligned)
    char* p = (char*)d_ws;
    f16* wq16 = (f16*)p; p += (size_t)QKVN * DIM * 2;
    f16* wo16 = (f16*)p; p += (size_t)DIM * DIM * 2;
    f16* wf1  = (f16*)p; p += (size_t)MLP_HIDN * DIM * 2;
    f16* wf2  = (f16*)p; p += (size_t)DIM * MLP_HIDN * 2;
    f16* act1 = (f16*)p; p += (size_t)MTOT * DIM * 2;      // also reused as h2
    f16* qkvb = (f16*)p; p += (size_t)MTOT * QKVN * 2;     // hid aliases qkvb+ocb
    f16* ocb  = (f16*)p; p += (size_t)MTOT * DIM * 2;
    float* x2 = (float*)p;
    f16* h2  = act1;
    f16* hid = qkvb;   // spans qkvb..ocb end: MTOT*1536 halves, both dead by then

    castw_kernel<<<(QKVN * DIM + 255) / 256, 256, 0, stream>>>(qkv_w, wq16, QKVN * DIM);
    castw_kernel<<<(DIM * DIM + 255) / 256, 256, 0, stream>>>(out_w, wo16, DIM * DIM);
    castw_kernel<<<(MLP_HIDN * DIM + 255) / 256, 256, 0, stream>>>(fc1_w, wf1, MLP_HIDN * DIM);
    castw_kernel<<<(DIM * MLP_HIDN + 255) / 256, 256, 0, stream>>>(fc2_w, wf2, DIM * MLP_HIDN);

    ln1win_kernel<<<MTOT / 8, 256, 0, stream>>>(x, n1w, n1b, act1);
    gemm_kernel<0><<<dim3(MTOT / 128, QKVN / 64), 256, 0, stream>>>(act1, wq16, qkv_b, DIM, QKVN, qkvb, nullptr, nullptr);
    attn_kernel<<<dim3(NW, HEADS), 128, 0, stream>>>(qkvb, ocb);
    gemm_kernel<1><<<dim3(MTOT / 128, DIM / 64), 256, 0, stream>>>(ocb, wo16, out_b, DIM, DIM, nullptr, x2, x);
    ln2_kernel<<<MTOT / 8, 256, 0, stream>>>(x2, n2w, n2b, h2);
    gemm_kernel<2><<<dim3(MTOT / 128, MLP_HIDN / 64), 256, 0, stream>>>(h2, wf1, fc1_b, DIM, MLP_HIDN, hid, nullptr, nullptr);
    gemm_kernel<3><<<dim3(MTOT / 128, DIM / 64), 256, 0, stream>>>(hid, wf2, fc2_b, MLP_HIDN, DIM, nullptr, out, x2);
}
